// WindowAttention_66571993089019
// MI455X (gfx1250) — matrix operand order repaired and checked by reading, not compiled, not run on hardware
//
#include <hip/hip_runtime.h>
#include <hip/hip_bf16.h>

typedef __bf16 bf16;
typedef bf16 v16bf __attribute__((ext_vector_type(16)));
typedef bf16 v8bf  __attribute__((ext_vector_type(8)));
typedef bf16 v4bf  __attribute__((ext_vector_type(4)));
typedef float v8f  __attribute__((ext_vector_type(8)));
typedef float v4f  __attribute__((ext_vector_type(4)));
typedef unsigned int tdm_v4u __attribute__((ext_vector_type(4)));
typedef int tdm_v8i __attribute__((ext_vector_type(8)));
typedef int tdm_v4i __attribute__((ext_vector_type(4)));

#define DIM    256
#define HEADS  8
#define HDIM   32
#define N1     50
#define SCALE  0.17677669529663687f

// LDS strides (elements)
#define XS   264   // xc / attn_out bf16 row stride (64 rows)
#define QS   776   // qkv bf16 row stride (64 rows, 768 cols)
#define VTS  72    // vT bf16 row stride (per head: 32 rows of 64 tokens)
#define WTS  264   // staged transposed weight chunk bf16 (256 n-rows x 264)
#define SCS  68    // scores f32 row stride (64 rows)
#define PS   72    // probs bf16 row stride (64 rows)

// LDS offsets (bytes), all 16B-aligned
#define XC_OFF    0                 // 64*264*2 = 33792   (reused as attn_out)
#define QKV_OFF   33792             // 64*776*2 = 99328
#define VT_OFF    133120            // 8*32*72*2 = 36864
#define WS_OFF    169984            // max(wT 135168, scores 8*17408=139264)
#define SMEM_BYTES 309248

// Workspace: 4 pre-converted bf16 weight images in LDS layout (n-major,
// k-contiguous, stride WTS): qkv chunks 0..2, then proj. 264*256 elems each.
#define WIMG_ELEMS  (256 * WTS)     // 67584 bf16 = 135168 bytes per image

__device__ inline v8bf ld8(const bf16* p) { return *(const v8bf*)p; }

// A operand 16x32 bf16: lane l holds row m=l&15; elems 0..7 = k in [8*half,+8),
// elems 8..15 = k in [16+8*half,+8). Two contiguous 16B LDS reads.
__device__ inline v16bf load_a(const bf16* base, int stride, int row0, int col0, int lane) {
  int m = lane & 15, half = lane >> 4;
  const bf16* p = base + (row0 + m) * stride + col0 + 8 * half;
  v8bf lo = ld8(p);
  v8bf hi = ld8(p + 16);
  return __builtin_shufflevector(lo, hi, 0,1,2,3,4,5,6,7,8,9,10,11,12,13,14,15);
}

// B operand 32x16 bf16 (k-contiguous, n-major buffer):
// lane l holds col n=l&15; elems 0..7 = k in [8*half,+8),
// elems 8..15 = k in [16+8*half,+8). Two contiguous 16B reads.
__device__ inline v16bf load_b(const bf16* baseT, int stride, int nrow0, int k0, int lane) {
  int n = lane & 15, half = lane >> 4;
  const bf16* p = baseT + (nrow0 + n) * stride + k0 + 8 * half;
  v8bf lo = ld8(p);
  v8bf hi = ld8(p + 16);
  return __builtin_shufflevector(lo, hi, 0,1,2,3,4,5,6,7,8,9,10,11,12,13,14,15);
}

__device__ inline v8f wmma_bf16(v16bf a, v16bf b, v8f c) {
  return __builtin_amdgcn_wmma_f32_16x16x32_bf16(false, a, false, b, (short)0, c, false, false);
}

// TDM: DMA one 256x264 bf16 weight image from global into LDS at WS_OFF.
// D# group0: count=1 | lds_addr | global_addr | type=2
// D# group1: data_size=2B, tensor_dim0=264, tensor_dim1=256,
//            tile_dim0=264, tile_dim1=256, tensor_dim0_stride=264
__device__ inline void tdm_load_wT(const bf16* src) {
  unsigned long long ga = (unsigned long long)(uintptr_t)src;
  tdm_v4u g0 = { 1u,                                  // count=1, user desc
                 (unsigned)WS_OFF,                    // lds_addr (bytes)
                 (unsigned)(ga & 0xFFFFFFFFu),        // global_addr[31:0]
                 (unsigned)((ga >> 32) | (2u << 30)) };// global_addr[56:32] | type=2
  tdm_v8i g1 = { (int)0x00010000,                     // data_size=1 (2 bytes)
                 (int)0x01080000,                     // tensor_dim0[15:0]=264 << 16
                 (int)0x01000000,                     // dim0 hi=0 | tensor_dim1 lo=256
                 (int)0x01080000,                     // dim1 hi=0 | tile_dim0=264
                 (int)0x00000100,                     // tile_dim1=256 | tile_dim2=0
                 (int)264,                            // tensor_dim0_stride lo32
                 0, 0 };                              // stride hi, dim1_stride
  tdm_v4i z4 = { 0, 0, 0, 0 };
  tdm_v8i z8 = { 0, 0, 0, 0, 0, 0, 0, 0 };
  __builtin_amdgcn_tensor_load_to_lds(g0, g1, z4, z4, z8, 0);
}

// ---- Pre-kernel: convert f32 weights to bf16 ws images (run once/launch) ----
__global__ __launch_bounds__(256)
void wa_convert_weights(const float* __restrict__ qkvw,
                        const float* __restrict__ projw,
                        bf16* __restrict__ ws) {
  int idx = blockIdx.x * 256 + threadIdx.x;           // 4*65536 total
  if (idx >= 4 * 65536) return;
  int img = idx >> 16, e = idx & 65535;
  int k = e >> 8, nn = e & 255;                       // consecutive tid -> nn (coalesced read)
  float v = (img < 3) ? qkvw[k * 768 + img * 256 + nn]
                      : projw[k * 256 + nn];
  ws[(size_t)img * WIMG_ELEMS + nn * WTS + k] = (bf16)v;
}

__global__ __launch_bounds__(256)
void WindowAttention_66571993089019_kernel(
    const float* __restrict__ x,        // (B, 49, 256)
    const float* __restrict__ mask,     // (64, 50, 50)
    const float* __restrict__ cls,      // (256)
    const float* __restrict__ qkvb,     // (768)
    const float* __restrict__ projb,    // (256)
    const bf16*  __restrict__ wsw,      // 4 bf16 weight images (LDS layout)
    float* __restrict__ out)            // [B*256 cls] ++ [B*49*256 rest]
{
  extern __shared__ char smem[];
  bf16*  xc  = (bf16*)(smem + XC_OFF);    // 64 x 256 (stride XS); later attn_out
  bf16*  qkv = (bf16*)(smem + QKV_OFF);   // 64 x 768 (stride QS); Q pre-scaled
  bf16*  vt  = (bf16*)(smem + VT_OFF);    // per head: 32 x 64 (stride VTS)
  bf16*  wT  = (bf16*)(smem + WS_OFF);    // 256 x 264 transposed weights

  const int tid  = threadIdx.x;
  const int lane = tid & 31;
  const int wave = tid >> 5;
  const int b    = blockIdx.x;
  const int n    = lane & 15, half = lane >> 4;

  // Kick off the TDM DMA for qkv chunk 0 immediately (overlaps x staging).
  if (wave == 0) tdm_load_wT(wsw);

  // ---- Phase 0: xc = [cls ; x[b]] as bf16, zero rows 50..63 ----
  for (int idx = tid; idx < 16 * 256; idx += 256) {
    int row = idx >> 6, col4 = (idx & 63) * 4;
    v4f v = {};
    if (row == 0)      v = *(const v4f*)(cls + col4);
    else if (row < N1) v = __builtin_nontemporal_load(
                             (const v4f*)(x + ((size_t)b * 49 + (row - 1)) * DIM + col4));
    v4bf o = { (bf16)v.x, (bf16)v.y, (bf16)v.z, (bf16)v.w };
    *(v4bf*)(xc + row * XS + col4) = o;               // ds_store_b64
  }
  for (int idx = tid; idx < 48 * 64; idx += 256) {
    int row = 16 + (idx >> 6), col4 = (idx & 63) * 4;
    v4f v = {};
    if (row < N1) v = __builtin_nontemporal_load(
                        (const v4f*)(x + ((size_t)b * 49 + (row - 1)) * DIM + col4));
    v4bf o = { (bf16)v.x, (bf16)v.y, (bf16)v.z, (bf16)v.w };
    *(v4bf*)(xc + row * XS + col4) = o;
  }
  if (wave == 0) __builtin_amdgcn_s_wait_tensorcnt(0);
  __syncthreads();

  // A tiles of xc for this wave's fixed M-tile: invariant across all 3 chunks
  const int mt = wave & 3;
  const int ntb = wave >> 2;        // nt = ntb + 2*t covers 0..15
  v16bf a[8];
  #pragma unroll
  for (int ks = 0; ks < 8; ++ks) a[ks] = load_a(xc, XS, mt * 16, ks * 32, lane);

  // ---- Phase 1: qkv = xc @ qkv_w + b (3 chunks of 256 cols) ----
  for (int c = 0; c < 3; ++c) {
    const float mul = (c == 0) ? SCALE : 1.0f;
    for (int t = 0; t < 8; ++t) {
      int nt = ntb + 2 * t;
      v16bf bt[8];
      #pragma unroll
      for (int j = 0; j < 8; ++j) bt[j] = load_b(wT, WTS, nt * 16, j * 32, lane);
      v8f acc = {};
      #pragma unroll
      for (int j = 0; j < 8; ++j) acc = wmma_bf16(a[j], bt[j], acc);

      int colc = nt * 16 + n;
      int colg = c * 256 + colc;
      float bias = qkvb[colg];
      bf16* qp = qkv + (mt * 16 + 8 * half) * QS + colg;
      #pragma unroll
      for (int r = 0; r < 8; ++r) qp[r * QS] = (bf16)((acc[r] + bias) * mul);
      if (c == 2) {                                   // V transposed per head
        int h2 = colc >> 5, d = colc & 31;
        bf16* vp = vt + h2 * (32 * VTS) + d * VTS + mt * 16 + 8 * half;
        #pragma unroll
        for (int r = 0; r < 8; ++r) vp[r] = (bf16)(acc[r] + bias);
      }
    }
    __syncthreads();                                  // everyone done with wT
    if (c < 2) {
      if (wave == 0) { tdm_load_wT(wsw + (size_t)(c + 1) * WIMG_ELEMS);
                       __builtin_amdgcn_s_wait_tensorcnt(0); }
      __syncthreads();
    }
  }

  // ---- Phase 2: per-head attention (wave h owns head h) ----
  const int h = wave;
  float* sc = (float*)(smem + WS_OFF + h * 17408);    // 64x64 f32 (stride SCS)
  bf16*  pb = (bf16*)(smem + WS_OFF + h * 17408);     // overlays sc, row-order safe
  const float* mwin = mask + (size_t)(b & 63) * (N1 * N1);

  {
    v16bf qa[4], kb[4];
    #pragma unroll
    for (int i = 0; i < 4; ++i) qa[i] = load_a(qkv, QS, i * 16, h * HDIM, lane);
    #pragma unroll
    for (int i = 0; i < 4; ++i) kb[i] = load_b(qkv + 256 + h * HDIM, QS, i * 16, 0, lane);
    #pragma unroll
    for (int mi = 0; mi < 4; ++mi) {
      #pragma unroll
      for (int ni = 0; ni < 4; ++ni) {
        v8f acc = {};
        acc = wmma_bf16(qa[mi], kb[ni], acc);
        int gn = ni * 16 + n;
        float* sp = sc + (mi * 16 + 8 * half) * SCS + gn;
        #pragma unroll
        for (int r = 0; r < 8; ++r) {
          int gm = mi * 16 + r + 8 * half;
          float v = acc[r];
          if (gn >= N1)      v = -1e30f;              // padded keys -> prob 0
          else if (gm < N1)  v += mwin[gm * N1 + gn];
          sp[r * SCS] = v;
        }
      }
    }
  }

  // softmax rows (lane covers n=lane and n=lane+32), write bf16 probs
  for (int m = 0; m < 64; ++m) {
    float s0 = sc[m * SCS + lane];
    float s1 = sc[m * SCS + 32 + lane];
    float mx = fmaxf(s0, s1);
    #pragma unroll
    for (int off = 16; off; off >>= 1) mx = fmaxf(mx, __shfl_xor(mx, off, 32));
    float e0 = exp2f((s0 - mx) * 1.44269504f);
    float e1 = exp2f((s1 - mx) * 1.44269504f);
    float sum = e0 + e1;
    #pragma unroll
    for (int off = 16; off; off >>= 1) sum += __shfl_xor(sum, off, 32);
    float inv = 1.0f / sum;
    pb[m * PS + lane]      = (bf16)(e0 * inv);
    pb[m * PS + 32 + lane] = (bf16)(e1 * inv);
  }

  // O_h = P @ V_h  (M=64, N=32, K=64) -> attn_out cols [h*32, h*32+32)
  {
    const bf16* vth = vt + h * (32 * VTS);
    v16bf vb[2][2];
    #pragma unroll
    for (int ni = 0; ni < 2; ++ni)
      #pragma unroll
      for (int ks = 0; ks < 2; ++ks)
        vb[ni][ks] = load_b(vth, VTS, ni * 16, ks * 32, lane);
    for (int mi = 0; mi < 4; ++mi) {
      v16bf pa[2];
      #pragma unroll
      for (int ks = 0; ks < 2; ++ks) pa[ks] = load_a(pb, PS, mi * 16, ks * 32, lane);
      #pragma unroll
      for (int ni = 0; ni < 2; ++ni) {
        v8f acc = {};
        #pragma unroll
        for (int ks = 0; ks < 2; ++ks) acc = wmma_bf16(pa[ks], vb[ni][ks], acc);
        bf16* op = xc + (mi * 16 + 8 * half) * XS + h * HDIM + ni * 16 + n;
        #pragma unroll
        for (int r = 0; r < 8; ++r) op[r * XS] = (bf16)acc[r];
      }
    }
  }
  __syncthreads();                                    // P region free, attn_out ready

  // ---- Phase 3: out = attn_out @ proj_w + proj_b ----
  if (wave == 0) { tdm_load_wT(wsw + (size_t)3 * WIMG_ELEMS);
                   __builtin_amdgcn_s_wait_tensorcnt(0); }
  __syncthreads();
  v16bf oa[8];
  #pragma unroll
  for (int ks = 0; ks < 8; ++ks) oa[ks] = load_a(xc, XS, mt * 16, ks * 32, lane);
  const size_t cls_total = (size_t)gridDim.x * DIM;
  for (int t = 0; t < 8; ++t) {
    int nt = ntb + 2 * t;
    v16bf bt[8];
    #pragma unroll
    for (int j = 0; j < 8; ++j) bt[j] = load_b(wT, WTS, nt * 16, j * 32, lane);
    v8f acc = {};
    #pragma unroll
    for (int j = 0; j < 8; ++j) acc = wmma_bf16(oa[j], bt[j], acc);

    int col = nt * 16 + n;
    float bias = projb[col];
    #pragma unroll
    for (int r = 0; r < 8; ++r) {
      int m = mt * 16 + r + 8 * half;
      if (m < N1) {
        float v = acc[r] + bias;
        if (m == 0)
          __builtin_nontemporal_store(v, out + (size_t)b * DIM + col);
        else
          __builtin_nontemporal_store(v, out + cls_total + ((size_t)b * 49 + (m - 1)) * DIM + col);
      }
    }
  }
}

extern "C" void kernel_launch(void* const* d_in, const int* in_sizes, int n_in,
                              void* d_out, int out_size, void* d_ws, size_t ws_size,
                              hipStream_t stream) {
  const float* x     = (const float*)d_in[0];
  const float* mask  = (const float*)d_in[1];
  const float* cls   = (const float*)d_in[2];
  const float* qkvw  = (const float*)d_in[3];
  const float* qkvb  = (const float*)d_in[4];
  const float* projw = (const float*)d_in[5];
  const float* projb = (const float*)d_in[6];
  float* out = (float*)d_out;
  bf16* wsw  = (bf16*)d_ws;           // 4 * 135168 B = 540672 B used

  int B = in_sizes[0] / (49 * DIM);   // 2048

  wa_convert_weights<<<(4 * 65536 + 255) / 256, 256, 0, stream>>>(qkvw, projw, wsw);

  (void)hipFuncSetAttribute((const void*)WindowAttention_66571993089019_kernel,
                            hipFuncAttributeMaxDynamicSharedMemorySize, SMEM_BYTES);
  WindowAttention_66571993089019_kernel<<<B, 256, SMEM_BYTES, stream>>>(
      x, mask, cls, qkvb, projb, wsw, out);
}
